// non_local_57260503990846
// MI455X (gfx1250) — matrix operand order repaired and checked by reading, not compiled, not run on hardware
//
#include <hip/hip_runtime.h>

#define CCH 512      // channels C
#define HS  128      // HSIZE
#define NN  10368    // D*H*W

typedef __bf16 bf16;
typedef __attribute__((ext_vector_type(16))) __bf16 v16bf;
typedef __attribute__((ext_vector_type(8)))  __bf16 v8bf;
typedef __attribute__((ext_vector_type(8)))  float  v8f;

static __device__ __forceinline__ bf16 f2bf(float f) { return (bf16)f; }

static __device__ __forceinline__ v8f wmma_bf16(v16bf a, v16bf b, v8f c) {
  // D = A(16x32 bf16) * B(32x16 bf16) + C(16x16 f32)
  return __builtin_amdgcn_wmma_f32_16x16x32_bf16(false, a, false, b, (short)0, c,
                                                 false, false);
}

// ---- fragment loaders (wave32, contiguous k order for both A and B) ------------------------------
// A 16x32 (MxK), element h of lane: M = m0 + (lane&15),
//   K = k0 + 16*((lane>>4)&1) + h   -> one contiguous 16-elem run.

static __device__ __forceinline__ v16bf loadA_f32(const float* __restrict__ src,
                                                  int ld, int m0, int k0, int lane) {
  const float* p = src + (size_t)(m0 + (lane & 15)) * ld + k0 + 16 * ((lane >> 4) & 1);
  float4 q0 = *(const float4*)(p + 0);
  float4 q1 = *(const float4*)(p + 4);
  float4 q2 = *(const float4*)(p + 8);
  float4 q3 = *(const float4*)(p + 12);
  v16bf a;
  a[0]=f2bf(q0.x);  a[1]=f2bf(q0.y);  a[2]=f2bf(q0.z);  a[3]=f2bf(q0.w);
  a[4]=f2bf(q1.x);  a[5]=f2bf(q1.y);  a[6]=f2bf(q1.z);  a[7]=f2bf(q1.w);
  a[8]=f2bf(q2.x);  a[9]=f2bf(q2.y);  a[10]=f2bf(q2.z); a[11]=f2bf(q2.w);
  a[12]=f2bf(q3.x); a[13]=f2bf(q3.y); a[14]=f2bf(q3.z); a[15]=f2bf(q3.w);
  return a;
}

static __device__ __forceinline__ v16bf loadA_bf16(const bf16* __restrict__ src,
                                                   int ld, int m0, int k0, int lane) {
  const bf16* p = src + (size_t)(m0 + (lane & 15)) * ld + k0 + 16 * ((lane >> 4) & 1);
  v8bf lo = *(const v8bf*)(p);
  v8bf hi = *(const v8bf*)(p + 8);
  v16bf a;
#pragma unroll
  for (int i = 0; i < 8; ++i) { a[i] = lo[i]; a[8 + i] = hi[i]; }
  return a;
}

// B 32x16 (KxN), element h of lane: N = n0 + (lane&15), K = k0 + 16*((lane>>4)&1) + h.
// Source is K-major (rows = K): per-h loads coalesce across the 16 lanes (64B runs).
static __device__ __forceinline__ v16bf loadB_kmajor_f32(const float* __restrict__ src,
                                                         int ld, int k0, int n0, int lane) {
  const float* p = src + (size_t)(k0 + 16 * ((lane >> 4) & 1)) * ld + n0 + (lane & 15);
  v16bf b;
#pragma unroll
  for (int h = 0; h < 16; ++h) b[h] = f2bf(p[(size_t)h * ld]);
  return b;
}

// B 32x16 where source is N-major (row n0+(lane&15) holds the K-run): per-lane
// contiguous 64B (f32) / 32B (bf16).
static __device__ __forceinline__ v16bf loadB_nmajor_f32(const float* __restrict__ src,
                                                         int ld, int n0, int k0, int lane) {
  const float* p = src + (size_t)(n0 + (lane & 15)) * ld + k0 + 16 * ((lane >> 4) & 1);
  float4 q0 = *(const float4*)(p + 0);
  float4 q1 = *(const float4*)(p + 4);
  float4 q2 = *(const float4*)(p + 8);
  float4 q3 = *(const float4*)(p + 12);
  v16bf b;
  b[0]=f2bf(q0.x);  b[1]=f2bf(q0.y);  b[2]=f2bf(q0.z);  b[3]=f2bf(q0.w);
  b[4]=f2bf(q1.x);  b[5]=f2bf(q1.y);  b[6]=f2bf(q1.z);  b[7]=f2bf(q1.w);
  b[8]=f2bf(q2.x);  b[9]=f2bf(q2.y);  b[10]=f2bf(q2.z); b[11]=f2bf(q2.w);
  b[12]=f2bf(q3.x); b[13]=f2bf(q3.y); b[14]=f2bf(q3.z); b[15]=f2bf(q3.w);
  return b;
}

static __device__ __forceinline__ v16bf loadB_nmajor_bf16(const bf16* __restrict__ src,
                                                          int ld, int n0, int k0, int lane) {
  const bf16* p = src + (size_t)(n0 + (lane & 15)) * ld + k0 + 16 * ((lane >> 4) & 1);
  v8bf lo = *(const v8bf*)(p);
  v8bf hi = *(const v8bf*)(p + 8);
  v16bf b;
#pragma unroll
  for (int i = 0; i < 8; ++i) { b[i] = lo[i]; b[8 + i] = hi[i]; }
  return b;
}

// ---- kernels --------------------------------------------------------------

__global__ void k_zero(float* __restrict__ t) {
  int i = blockIdx.x * blockDim.x + threadIdx.x;
  if (i < CCH * HS) t[i] = 0.0f;
}

// U = Wi*X + bi, V = Wj*X + bj  (K = 512). Each wave: 16(M) x 64(N), U and V fused.
// Outputs: Utb = U^T bf16 [N x 128], Vb = V bf16 [128 x N].
__global__ void k_uv(const float* __restrict__ X, const float* __restrict__ Wi,
                     const float* __restrict__ bi, const float* __restrict__ Wj,
                     const float* __restrict__ bj, bf16* __restrict__ Utb,
                     bf16* __restrict__ Vb) {
  const int lane = threadIdx.x & 31;
  const int unit = blockIdx.x * (blockDim.x >> 5) + (threadIdx.x >> 5); // 0..1295
  const int nb = unit % 162, mt = unit / 162;
  const int m0 = mt * 16, n0 = nb * 64;

  v8f zero = {0.f, 0.f, 0.f, 0.f, 0.f, 0.f, 0.f, 0.f};
  v8f accU[4], accV[4];
#pragma unroll
  for (int t = 0; t < 4; ++t) { accU[t] = zero; accV[t] = zero; }

  for (int k0 = 0; k0 < CCH; k0 += 32) {
    v16bf aU = loadA_f32(Wi, CCH, m0, k0, lane);
    v16bf aV = loadA_f32(Wj, CCH, m0, k0, lane);
#pragma unroll
    for (int t = 0; t < 4; ++t) {
      v16bf bb = loadB_kmajor_f32(X, NN, k0, n0 + 16 * t, lane);
      accU[t] = wmma_bf16(aU, bb, accU[t]);
      accV[t] = wmma_bf16(aV, bb, accV[t]);
    }
  }

  const int hi = (lane >> 4) & 1;
  const int nc = lane & 15;
  float bU[8], bV[8];
#pragma unroll
  for (int r = 0; r < 8; ++r) {
    int h = m0 + 8 * hi + r;
    bU[r] = bi[h];
    bV[r] = bj[h];
  }
#pragma unroll
  for (int t = 0; t < 4; ++t) {
    int n = n0 + 16 * t + nc;
    v8bf uo;
#pragma unroll
    for (int r = 0; r < 8; ++r) {
      uo[r] = f2bf(accU[t][r] + bU[r]);
      Vb[(size_t)(m0 + 8 * hi + r) * NN + n] = f2bf(accV[t][r] + bV[r]);
    }
    *(v8bf*)(Utb + (size_t)n * HS + m0 + 8 * hi) = uo;  // 16B packed store
  }
}

// T = V * X^T  (K = N = 10368), split-K 9 ways, atomic f32 accumulate into
// Tacc laid out as T^T: Tacc[c*128 + h].
__global__ void k_t(const float* __restrict__ X, const bf16* __restrict__ Vb,
                    float* __restrict__ Tacc) {
  const int lane = threadIdx.x & 31;
  const int unit = blockIdx.x * (blockDim.x >> 5) + (threadIdx.x >> 5); // 0..2303
  const int ks = unit % 9;
  const int ct = (unit / 9) % 32;
  const int mt = unit / (9 * 32);
  const int m0 = mt * 16, c0 = ct * 16;

  v8f acc = {0.f, 0.f, 0.f, 0.f, 0.f, 0.f, 0.f, 0.f};
  const int kbeg = ks * 36, kend = kbeg + 36;   // 324 = 9*36 k-steps of 32
  for (int kk = kbeg; kk < kend; ++kk) {
    int n0 = kk * 32;
    v16bf aV = loadA_bf16(Vb, NN, m0, n0, lane);           // rows of V
    v16bf bX = loadB_nmajor_f32(X, NN, c0, n0, lane);      // X^T cols = X rows
    acc = wmma_bf16(aV, bX, acc);
  }
  const int hi = (lane >> 4) & 1, cc = lane & 15;
#pragma unroll
  for (int r = 0; r < 8; ++r)
    atomicAdd(&Tacc[(size_t)(c0 + cc) * HS + m0 + 8 * hi + r], acc[r]);
}

__global__ void k_tcvt(const float* __restrict__ Tacc, bf16* __restrict__ Ttb) {
  int i = blockIdx.x * blockDim.x + threadIdx.x;
  if (i < CCH * HS) Ttb[i] = f2bf(Tacc[i]);
}

// out = X + scale * (T^T * U)  (K = 128). Each wave: 16(C) x 64(N).
__global__ void k_out(const float* __restrict__ X, const bf16* __restrict__ Ttb,
                      const bf16* __restrict__ Utb, const float* __restrict__ agg,
                      float* __restrict__ out) {
  const int lane = threadIdx.x & 31;
  const int unit = blockIdx.x * (blockDim.x >> 5) + (threadIdx.x >> 5); // 0..5183
  const int nb = unit % 162;
  const int cm = unit / 162;                   // 0..31
  const int c0 = cm * 16, n0 = nb * 64;

  v8f zero = {0.f, 0.f, 0.f, 0.f, 0.f, 0.f, 0.f, 0.f};
  v8f acc[4];
#pragma unroll
  for (int t = 0; t < 4; ++t) acc[t] = zero;

#pragma unroll
  for (int k0 = 0; k0 < HS; k0 += 32) {
    v16bf aT = loadA_bf16(Ttb, HS, c0, k0, lane);          // rows of T^T
#pragma unroll
    for (int t = 0; t < 4; ++t) {
      v16bf bU = loadB_nmajor_bf16(Utb, HS, n0 + 16 * t, k0, lane); // U cols = U^T rows
      acc[t] = wmma_bf16(aT, bU, acc[t]);
    }
  }

  const float scale = agg[0] / (float)NN;
  const int hi = (lane >> 4) & 1, nc = lane & 15;
#pragma unroll
  for (int t = 0; t < 4; ++t) {
    int n = n0 + 16 * t + nc;
#pragma unroll
    for (int r = 0; r < 8; ++r) {
      size_t idx = (size_t)(c0 + 8 * hi + r) * NN + n;
      out[idx] = X[idx] + scale * acc[t][r];
    }
  }
}

extern "C" void kernel_launch(void* const* d_in, const int* in_sizes, int n_in,
                              void* d_out, int out_size, void* d_ws, size_t ws_size,
                              hipStream_t stream) {
  (void)in_sizes; (void)n_in; (void)out_size; (void)ws_size;
  const float* X   = (const float*)d_in[0];
  const float* Wi  = (const float*)d_in[1];
  const float* bi  = (const float*)d_in[2];
  const float* Wj  = (const float*)d_in[3];
  const float* bj  = (const float*)d_in[4];
  const float* agg = (const float*)d_in[5];
  float* out = (float*)d_out;

  char* w = (char*)d_ws;                      // ~5.7 MB total
  bf16*  Utb  = (bf16*)w;  w += (size_t)NN * HS * sizeof(bf16);   // U^T [N x 128]
  bf16*  Vb   = (bf16*)w;  w += (size_t)HS * NN * sizeof(bf16);   // V   [128 x N]
  float* Tacc = (float*)w; w += (size_t)CCH * HS * sizeof(float); // T^T f32 accum
  bf16*  Ttb  = (bf16*)w;                                          // T^T bf16

  k_zero<<<(CCH * HS + 255) / 256, 256, 0, stream>>>(Tacc);
  k_uv  <<<162, 256, 0, stream>>>(X, Wi, bi, Wj, bj, Utb, Vb);     // 1296 waves
  k_t   <<<288, 256, 0, stream>>>(X, Vb, Tacc);                    // 2304 waves
  k_tcvt<<<(CCH * HS + 255) / 256, 256, 0, stream>>>(Tacc, Ttb);
  k_out <<<648, 256, 0, stream>>>(X, Ttb, Utb, agg, out);          // 5184 waves
}
